// PatchedDeepseekV2MoE_6047313953505
// MI455X (gfx1250) — matrix operand order repaired and checked by reading, not compiled, not run on hardware
//
#include <hip/hip_runtime.h>
#include <hip/hip_bf16.h>

// Problem constants (from reference setup_inputs)
#define T_TOK 1024
#define HDIM  2048
#define FDIM  1408
#define EDIM  32
#define KTOP  6
#define TK    (T_TOK * KTOP)   // 6144 token-slot pairs

typedef __attribute__((ext_vector_type(16))) __bf16 v16bf;
typedef __attribute__((ext_vector_type(8)))  float  v8f;

// ---- CDNA5 async global->LDS staging (gated: fall back to sync path) ------
#if __has_builtin(__builtin_amdgcn_global_load_async_to_lds_b128)
#define MOE_ASYNC_STAGE 1
#else
#define MOE_ASYNC_STAGE 0
#endif

#if __has_builtin(__builtin_amdgcn_s_wait_asynccnt)
#define WAIT_ASYNC(n) __builtin_amdgcn_s_wait_asynccnt(n)
#else
#define WAIT_ASYNC(n) asm volatile("s_wait_asynccnt %0" ::"i"(n) : "memory")
#endif

#if MOE_ASYNC_STAGE
// Builtin signature (from hipcc diagnostic): parameter 1 is
// 'int __attribute__((vector_size(16))) addrspace(1)*', parameter 2 the
// addrspace(3) analog, then imm offset and imm cpol.
typedef int moe_v4i __attribute__((vector_size(16)));
typedef __attribute__((address_space(1))) moe_v4i* moe_gp16;
typedef __attribute__((address_space(3))) moe_v4i* moe_lp16;

// Copy 16B global -> LDS via the async path (tracked by ASYNCcnt, bypasses
// VGPRs entirely: no global_load_b128 + ds_store_b128 round trip).
__device__ __forceinline__ void async_cp16(const float* g, float* l) {
    __builtin_amdgcn_global_load_async_to_lds_b128((moe_gp16)g, (moe_lp16)l, 0, 0);
}
#endif

// Convert 16 consecutive f32 (as 4 float4) to a bf16 WMMA operand fragment.
__device__ __forceinline__ v16bf cvt16(float4 a, float4 b, float4 c, float4 d) {
    v16bf r;
    r[0]  = (__bf16)a.x; r[1]  = (__bf16)a.y; r[2]  = (__bf16)a.z; r[3]  = (__bf16)a.w;
    r[4]  = (__bf16)b.x; r[5]  = (__bf16)b.y; r[6]  = (__bf16)b.z; r[7]  = (__bf16)b.w;
    r[8]  = (__bf16)c.x; r[9]  = (__bf16)c.y; r[10] = (__bf16)c.z; r[11] = (__bf16)c.w;
    r[12] = (__bf16)d.x; r[13] = (__bf16)d.y; r[14] = (__bf16)d.z; r[15] = (__bf16)d.w;
    return r;
}

// ---------------------------------------------------------------------------
// Kernel 1: deterministic routing. One wave (32 threads), thread e handles
// expert e: count, exclusive-scan, then place (token, weight) pairs in stable
// entry order. 32 lanes x 6144 iters -> trivial cost.
// ---------------------------------------------------------------------------
__global__ __launch_bounds__(32)
void moe_route_kernel(const long long* __restrict__ ids,
                      const float*     __restrict__ tw,
                      int* __restrict__ off, int* __restrict__ tokp,
                      float* __restrict__ wgtp) {
    __shared__ int cnts[EDIM];
    const int e = threadIdx.x;
    int c = 0;
    for (int i = 0; i < TK; ++i)
        if ((int)ids[i] == e) ++c;
    cnts[e] = c;
    __syncthreads();
    if (e == 0) {
        int s = 0;
        for (int j = 0; j < EDIM; ++j) { int t = cnts[j]; cnts[j] = s; off[j] = s; s += t; }
        off[EDIM] = s;   // == TK
    }
    __syncthreads();
    int pos = cnts[e];
    for (int i = 0; i < TK; ++i) {
        if ((int)ids[i] == e) {
            tokp[pos] = i / KTOP;   // token id
            wgtp[pos] = tw[i];      // routing weight
            ++pos;
        }
    }
}

// ---------------------------------------------------------------------------
// Kernel 2: GEMM1 + SwiGLU.  gu[r, f] = sum_h x[tok[r], h] * Wgu[e, f, h]
// 256 threads = 8 waves; wave w owns gate cols [nb, nb+16) AND paired up cols
// [F+nb, ...); 4 M-tiles (64 token rows) per block. Token tile double-buffered
// in LDS via async global->LDS copies so the next tile streams during WMMA.
// Grid: (F/128=11, E=32, 96 M-tiles worst case; extras early-exit).
// ---------------------------------------------------------------------------
__global__ __launch_bounds__(256)
void moe_gemm1_swiglu(const float* __restrict__ x,
                      const float* __restrict__ wgu,
                      const int*   __restrict__ off,
                      const int*   __restrict__ tokp,
                      float*       __restrict__ hbuf) {
    const int e     = blockIdx.y;
    const int r0    = off[e];
    const int cnt   = off[e + 1] - r0;
    const int mbase = blockIdx.z * 64;
    if (mbase >= cnt) return;

    const int tid  = threadIdx.x;
    const int wid  = tid >> 5;
    const int lane = tid & 31;
    const int nb   = blockIdx.x * 128 + wid * 16;   // gate column base, < F

    __shared__ float xs[2][64][36];  // 2 x (64 rows x 32 K), stride 36 (16B aligned, skewed)

    // LDS staging assignment: thread -> (row 0..63, 8-float chunk of K)
    const int srow  = tid >> 2;
    const int skk   = (tid & 3) * 8;
    const int sgrow = mbase + srow;
    const float* sx = nullptr;
    if (sgrow < cnt) sx = x + (size_t)tokp[r0 + sgrow] * HDIM;

    // WMMA lane roles (contiguous k order used here: K = 16*lhalf + j for both A and B)
    const int lhalf = lane >> 4;        // 0: lanes 0-15, 1: lanes 16-31
    const int ncol  = lane & 15;        // B/C/D column within tile
    const int koff  = lhalf * 16;       // B fragment: K = koff + j
    const int aoff  = lhalf * 8;        // A fragment: K = koff + j

    const size_t wrow_g = ((size_t)e * (2 * FDIM) + (nb + ncol)) * HDIM;
    const size_t wrow_u = ((size_t)e * (2 * FDIM) + (FDIM + nb + ncol)) * HDIM;

    v8f cg[4] = {v8f{}, v8f{}, v8f{}, v8f{}};
    v8f cu[4] = {v8f{}, v8f{}, v8f{}, v8f{}};

    const int NSTEP = HDIM / 32;   // 64

#if MOE_ASYNC_STAGE
    if (sx) {   // prologue: tile 0 -> buffer 0 (2 async b128 per thread)
        async_cp16(sx + skk,     &xs[0][srow][skk]);
        async_cp16(sx + skk + 4, &xs[0][srow][skk + 4]);
    }
#endif

    for (int s = 0; s < NSTEP; ++s) {
        const int k = s * 32;
#if MOE_ASYNC_STAGE
        const float (*xt)[36] = xs[s & 1];
        if (s + 1 < NSTEP) {
            if (sx) {   // issue tile s+1 into other buffer, then wait tile s
                async_cp16(sx + k + 32 + skk,     &xs[(s & 1) ^ 1][srow][skk]);
                async_cp16(sx + k + 32 + skk + 4, &xs[(s & 1) ^ 1][srow][skk + 4]);
            }
            WAIT_ASYNC(2);
        } else {
            WAIT_ASYNC(0);
        }
        __syncthreads();           // tile s visible to all waves
#else
        const float (*xt)[36] = xs[0];
        __syncthreads();           // previous readers done
        float4 z0 = make_float4(0.f, 0.f, 0.f, 0.f), z1 = z0;
        if (sx) {
            z0 = *(const float4*)(sx + k + skk);
            z1 = *(const float4*)(sx + k + skk + 4);
        }
        *(float4*)&xs[0][srow][skk]     = z0;
        *(float4*)&xs[0][srow][skk + 4] = z1;
        __syncthreads();
#endif

        const float* bgp = wgu + wrow_g + k + koff;
        const float* bup = wgu + wrow_u + k + koff;
        v16bf bg = cvt16(*(const float4*)(bgp),     *(const float4*)(bgp + 4),
                         *(const float4*)(bgp + 8), *(const float4*)(bgp + 12));
        v16bf bu = cvt16(*(const float4*)(bup),     *(const float4*)(bup + 4),
                         *(const float4*)(bup + 8), *(const float4*)(bup + 12));

#pragma unroll
        for (int mt = 0; mt < 4; ++mt) {
            const float* rp = &xt[mt * 16 + ncol][0];
            v16bf a = cvt16(*(const float4*)(rp + koff),     *(const float4*)(rp + koff + 4),
                            *(const float4*)(rp + koff + 8), *(const float4*)(rp + koff + 12));
            cg[mt] = __builtin_amdgcn_wmma_f32_16x16x32_bf16(false, a, false, bg,
                                                             (short)0, cg[mt], false, false);
            cu[mt] = __builtin_amdgcn_wmma_f32_16x16x32_bf16(false, a, false, bu,
                                                             (short)0, cu[mt], false, false);
        }
#if MOE_ASYNC_STAGE
        __syncthreads();           // all waves done with this buffer before refill
#endif
    }

    // SwiGLU epilogue: h = silu(gate) * up, write to workspace h buffer
    const int mofs = lhalf * 8;
#pragma unroll
    for (int mt = 0; mt < 4; ++mt) {
#pragma unroll
        for (int i = 0; i < 8; ++i) {
            const int rowg = mbase + mt * 16 + mofs + i;
            if (rowg < cnt) {
                const float g = cg[mt][i];
                const float u = cu[mt][i];
                const float s = g / (1.0f + __expf(-g));
                hbuf[(size_t)(r0 + rowg) * FDIM + (nb + ncol)] = s * u;
            }
        }
    }
}

// ---------------------------------------------------------------------------
// Kernel 3: GEMM2 + weighted scatter. y[r, n] = sum_f h[r, f] * Wd[e, n, f];
// out[tok[r], n] += w[r] * y[r, n]  via global_atomic_add_f32.
// Grid: (H/128=16, E=32, 96). Same double-buffered async LDS staging for h.
// ---------------------------------------------------------------------------
__global__ __launch_bounds__(256)
void moe_gemm2_down(const float* __restrict__ hbuf,
                    const float* __restrict__ wd,
                    const int*   __restrict__ off,
                    const int*   __restrict__ tokp,
                    const float* __restrict__ wgtp,
                    float*       __restrict__ out) {
    const int e     = blockIdx.y;
    const int r0    = off[e];
    const int cnt   = off[e + 1] - r0;
    const int mbase = blockIdx.z * 64;
    if (mbase >= cnt) return;

    const int tid  = threadIdx.x;
    const int wid  = tid >> 5;
    const int lane = tid & 31;
    const int nb   = blockIdx.x * 128 + wid * 16;   // output hidden-dim column base

    __shared__ float hs[2][64][36];

    const int srow  = tid >> 2;
    const int skk   = (tid & 3) * 8;
    const int sgrow = mbase + srow;
    const float* sh = (sgrow < cnt) ? (hbuf + (size_t)(r0 + sgrow) * FDIM) : nullptr;

    const int lhalf = lane >> 4;
    const int ncol  = lane & 15;
    const int koff  = lhalf * 16;
    const int aoff  = lhalf * 8;

    const size_t wrow = ((size_t)e * HDIM + (nb + ncol)) * FDIM;

    v8f c[4] = {v8f{}, v8f{}, v8f{}, v8f{}};

    const int NSTEP = FDIM / 32;   // 44

#if MOE_ASYNC_STAGE
    if (sh) {
        async_cp16(sh + skk,     &hs[0][srow][skk]);
        async_cp16(sh + skk + 4, &hs[0][srow][skk + 4]);
    }
#endif

    for (int s = 0; s < NSTEP; ++s) {
        const int kf = s * 32;
#if MOE_ASYNC_STAGE
        const float (*ht)[36] = hs[s & 1];
        if (s + 1 < NSTEP) {
            if (sh) {
                async_cp16(sh + kf + 32 + skk,     &hs[(s & 1) ^ 1][srow][skk]);
                async_cp16(sh + kf + 32 + skk + 4, &hs[(s & 1) ^ 1][srow][skk + 4]);
            }
            WAIT_ASYNC(2);
        } else {
            WAIT_ASYNC(0);
        }
        __syncthreads();
#else
        const float (*ht)[36] = hs[0];
        __syncthreads();
        float4 z0 = make_float4(0.f, 0.f, 0.f, 0.f), z1 = z0;
        if (sh) {
            z0 = *(const float4*)(sh + kf + skk);
            z1 = *(const float4*)(sh + kf + skk + 4);
        }
        *(float4*)&hs[0][srow][skk]     = z0;
        *(float4*)&hs[0][srow][skk + 4] = z1;
        __syncthreads();
#endif

        const float* bp = wd + wrow + kf + koff;
        v16bf b = cvt16(*(const float4*)(bp),     *(const float4*)(bp + 4),
                        *(const float4*)(bp + 8), *(const float4*)(bp + 12));

#pragma unroll
        for (int mt = 0; mt < 4; ++mt) {
            const float* rp = &ht[mt * 16 + ncol][0];
            v16bf a = cvt16(*(const float4*)(rp + koff),     *(const float4*)(rp + koff + 4),
                            *(const float4*)(rp + koff + 8), *(const float4*)(rp + koff + 12));
            c[mt] = __builtin_amdgcn_wmma_f32_16x16x32_bf16(false, a, false, b,
                                                            (short)0, c[mt], false, false);
        }
#if MOE_ASYNC_STAGE
        __syncthreads();
#endif
    }

    // Weighted scatter-accumulate into out (each token hit by K=6 experts)
    const int mofs = lhalf * 8;
#pragma unroll
    for (int mt = 0; mt < 4; ++mt) {
#pragma unroll
        for (int i = 0; i < 8; ++i) {
            const int rowg = mbase + mt * 16 + mofs + i;
            if (rowg < cnt) {
                const int   tkr = tokp[r0 + rowg];
                const float w   = wgtp[r0 + rowg];
                atomicAdd(out + (size_t)tkr * HDIM + (nb + ncol), c[mt][i] * w);
            }
        }
    }
}

// ---------------------------------------------------------------------------
// Launch
// ---------------------------------------------------------------------------
extern "C" void kernel_launch(void* const* d_in, const int* in_sizes, int n_in,
                              void* d_out, int out_size, void* d_ws, size_t ws_size,
                              hipStream_t stream) {
    const float*     x   = (const float*)d_in[0];       // (T, H)
    const float*     wgu = (const float*)d_in[1];       // (E, 2F, H)
    const float*     wd  = (const float*)d_in[2];       // (E, H, F)
    const float*     tw  = (const float*)d_in[3];       // (T, K)
    const long long* ids = (const long long*)d_in[4];   // (T, K) int64
    float*           out = (float*)d_out;               // (T, H)

    // Workspace layout (floats/int32, 16B-aligned sections):
    //   [0,64)                 : expert offsets (E+1 used)
    //   [64, 64+TK)            : routed token ids
    //   [64+TK, 64+2*TK)       : routed weights
    //   [64+2*TK, ...)         : h buffer, TK x F floats (~34.6 MB)
    int*   off  = (int*)d_ws;
    int*   tokp = off + 64;
    float* wgtp = (float*)(tokp + TK);
    float* hbuf = wgtp + TK;

    (void)hipMemsetAsync(d_out, 0, (size_t)out_size * sizeof(float), stream);

    moe_route_kernel<<<1, 32, 0, stream>>>(ids, tw, off, tokp, wgtp);

    moe_gemm1_swiglu<<<dim3(FDIM / 128, EDIM, 96), 256, 0, stream>>>(
        x, wgu, off, tokp, hbuf);

    moe_gemm2_down<<<dim3(HDIM / 128, EDIM, 96), 256, 0, stream>>>(
        hbuf, wd, off, tokp, wgtp, out);
}
